// ElasticMLPTriton_63591285785172
// MI455X (gfx1250) — matrix operand order repaired and checked by reading, not compiled, not run on hardware
//
#include <hip/hip_runtime.h>

// CDNA5 / gfx1250: wave32 WMMA, bf16x3 fp32-emulation MLP.
// out = relu(h W^T + b) (+x for layer>0), 8 layers, HIDDEN=128, BATCH=262144.
// Pre-pass splits fp32 weights into bf16 hi/lo planes in d_ws so the hot loop
// has (almost) no conversion VALU and feeds v_wmma_f32_16x16x32_bf16 directly.

typedef __attribute__((ext_vector_type(16))) __bf16 v16bf;
typedef __attribute__((ext_vector_type(8)))  __bf16 v8bf;
typedef __attribute__((ext_vector_type(8)))  float  v8f;

#define NUM_LAYERS 8
#define HIDDEN     128
#define TM         64    // batch rows per workgroup
#define MT         (TM / 16)
#define LDS_S      136   // padded LDS row stride (bf16 elems): 272B rows -> conflict-free A reads
#define W_ELEMS    (NUM_LAYERS * HIDDEN * HIDDEN)   // 131072

__device__ __forceinline__ void split_bf16(float f, __bf16& hi, __bf16& lo) {
    hi = (__bf16)f;                 // RNE to bf16
    lo = (__bf16)(f - (float)hi);   // residual, also bf16
}

// ---- pre-pass: split fp32 weights into bf16 hi/lo planes (one-shot, 512KB) ----
__global__ __launch_bounds__(256)
void split_weights_kernel(const float* __restrict__ w,
                          __bf16* __restrict__ whi,
                          __bf16* __restrict__ wlo)
{
    int idx = blockIdx.x * 256 + threadIdx.x;   // grid sized to exactly W_ELEMS
    float f = w[idx];
    __bf16 h, l; split_bf16(f, h, l);
    whi[idx] = h;
    wlo[idx] = l;
}

template <bool PRESPLIT>
__global__ __launch_bounds__(256)
void elastic_mlp_wmma_kernel(const float* __restrict__ x,
                             const float* __restrict__ w,
                             const __bf16* __restrict__ whi,
                             const __bf16* __restrict__ wlo,
                             const float* __restrict__ b,
                             float* __restrict__ out)
{
    __shared__ __attribute__((aligned(16))) __bf16 s_hi[TM * LDS_S];
    __shared__ __attribute__((aligned(16))) __bf16 s_lo[TM * LDS_S];

    const int tid  = threadIdx.x;
    const int wave = tid >> 5;          // 0..7 -> N strip of 16 output columns
    const int lane = tid & 31;
    const int half = lane >> 4;         // lane group 0/1
    const int l16  = lane & 15;
    const int row_base = blockIdx.x * TM;
    const int j = wave * 16 + l16;      // output column owned by this lane (B/C/D col)

    // ---- stage input tile into LDS as bf16 hi/lo planes (coalesced) ----
    #pragma unroll
    for (int i = 0; i < (TM * HIDDEN) / 256; ++i) {
        int idx = tid + 256 * i;
        int r = idx >> 7, c = idx & 127;
        float f = x[(size_t)(row_base + r) * HIDDEN + c];
        __bf16 h, l; split_bf16(f, h, l);
        s_hi[r * LDS_S + c] = h;
        s_lo[r * LDS_S + c] = l;
    }

    // ---- preload skip values in C/D accumulator layout (lane 0-15: M=r, 16-31: M=r+8) ----
    v8f xskip[MT];
    #pragma unroll
    for (int m = 0; m < MT; ++m) {
        #pragma unroll
        for (int r = 0; r < 8; ++r) {
            xskip[m][r] = x[(size_t)(row_base + m * 16 + r + 8 * half) * HIDDEN + j];
        }
    }

    __syncthreads();

    for (int layer = 0; layer < NUM_LAYERS; ++layer) {
        const float bias = b[layer * HIDDEN + j];

        // ---- this lane's B operands: B[k][n] = w[j][k]; this code's k order (contiguous, same as A):
        // lanes 0-15 hold K=0..15 (elem e == K), lanes 16-31 hold K=16..31,
        // so each lane reads 16 contiguous K values at k0 = 32c + 16*half.
        v16bf bhi[4], blo[4];
        if (PRESPLIT) {
            const __bf16* ph = whi + (size_t)layer * HIDDEN * HIDDEN + (size_t)j * HIDDEN;
            const __bf16* pl = wlo + (size_t)layer * HIDDEN * HIDDEN + (size_t)j * HIDDEN;
            #pragma unroll
            for (int c = 0; c < 4; ++c) {
                const int k0 = 32 * c + 16 * half;
                v8bf h0 = *(const v8bf*)(ph + k0);
                v8bf h1 = *(const v8bf*)(ph + k0 + 8);
                v8bf l0 = *(const v8bf*)(pl + k0);
                v8bf l1 = *(const v8bf*)(pl + k0 + 8);
                bhi[c] = __builtin_shufflevector(h0, h1, 0,1,2,3,4,5,6,7,8,9,10,11,12,13,14,15);
                blo[c] = __builtin_shufflevector(l0, l1, 0,1,2,3,4,5,6,7,8,9,10,11,12,13,14,15);
            }
        } else {
            const float* wl = w + (size_t)layer * HIDDEN * HIDDEN + (size_t)j * HIDDEN;
            #pragma unroll
            for (int c = 0; c < 4; ++c) {
                const float* p = wl + 32 * c + 16 * half;
                float4 f0 = ((const float4*)p)[0];
                float4 f1 = ((const float4*)p)[1];
                float4 f2 = ((const float4*)p)[2];
                float4 f3 = ((const float4*)p)[3];
                float fv[16] = {f0.x,f0.y,f0.z,f0.w, f1.x,f1.y,f1.z,f1.w,
                                f2.x,f2.y,f2.z,f2.w, f3.x,f3.y,f3.z,f3.w};
                #pragma unroll
                for (int e = 0; e < 16; ++e) {
                    __bf16 h, l; split_bf16(fv[e], h, l);
                    bhi[c][e] = h; blo[c][e] = l;
                }
            }
        }

        // ---- accumulate: D = A x B, A from LDS (16x32 bf16 layout, contiguous 16-K halves) ----
        v8f acc[MT];
        #pragma unroll
        for (int m = 0; m < MT; ++m) {
            v8f a = {};
            #pragma unroll
            for (int c = 0; c < 4; ++c) {
                // A lane layout: row M = l16; half h: K = 16h+e (contiguous, matches B)
                const int e0 = (m * 16 + l16) * LDS_S + 32 * c + 16 * half;
                v8bf h0 = *(const v8bf*)(s_hi + e0);        // K = k0 .. k0+7
                v8bf h1 = *(const v8bf*)(s_hi + e0 + 8);    // K = k0+8 .. k0+15
                v8bf l0 = *(const v8bf*)(s_lo + e0);
                v8bf l1 = *(const v8bf*)(s_lo + e0 + 8);
                v16bf ah = __builtin_shufflevector(h0, h1, 0,1,2,3,4,5,6,7,8,9,10,11,12,13,14,15);
                v16bf al = __builtin_shufflevector(l0, l1, 0,1,2,3,4,5,6,7,8,9,10,11,12,13,14,15);
                // bf16x3 emulation: hi*hi + hi*lo + lo*hi, fp32 accumulate
                a = __builtin_amdgcn_wmma_f32_16x16x32_bf16(false, ah, false, bhi[c], (short)0, a, false, false);
                a = __builtin_amdgcn_wmma_f32_16x16x32_bf16(false, ah, false, blo[c], (short)0, a, false, false);
                a = __builtin_amdgcn_wmma_f32_16x16x32_bf16(false, al, false, bhi[c], (short)0, a, false, false);
            }
            acc[m] = a;
        }

        __syncthreads();   // all waves done reading this layer's activations

        if (layer == NUM_LAYERS - 1) {
            // epilogue: bias + relu + skip, store fp32 to global (contiguous 64B per half-wave)
            #pragma unroll
            for (int m = 0; m < MT; ++m) {
                #pragma unroll
                for (int r = 0; r < 8; ++r) {
                    float h = fmaxf(acc[m][r] + bias, 0.0f) + xskip[m][r];
                    out[(size_t)(row_base + m * 16 + r + 8 * half) * HIDDEN + j] = h;
                }
            }
        } else {
            // epilogue: bias + relu (+ skip for layer>0), split and write next activations
            #pragma unroll
            for (int m = 0; m < MT; ++m) {
                #pragma unroll
                for (int r = 0; r < 8; ++r) {
                    float h = fmaxf(acc[m][r] + bias, 0.0f);
                    if (layer > 0) h += xskip[m][r];
                    __bf16 hh, ll; split_bf16(h, hh, ll);
                    int e = (m * 16 + r + 8 * half) * LDS_S + j;
                    s_hi[e] = hh;
                    s_lo[e] = ll;
                }
            }
            __syncthreads();   // writes visible before next layer's reads
        }
    }
}

extern "C" void kernel_launch(void* const* d_in, const int* in_sizes, int n_in,
                              void* d_out, int out_size, void* d_ws, size_t ws_size,
                              hipStream_t stream) {
    const float* x  = (const float*)d_in[0];  // (BATCH, 128) fp32
    const float* w  = (const float*)d_in[1];  // (8, 128, 128) fp32
    const float* bb = (const float*)d_in[2];  // (8, 128) fp32
    float* out = (float*)d_out;               // (BATCH, 128) fp32

    const int batch = in_sizes[0] / HIDDEN;
    dim3 grid(batch / TM);
    dim3 block(256);

    const size_t need = (size_t)W_ELEMS * 2 * sizeof(__bf16);   // hi + lo planes = 512KB
    if (ws_size >= need) {
        __bf16* whi = (__bf16*)d_ws;
        __bf16* wlo = whi + W_ELEMS;
        split_weights_kernel<<<W_ELEMS / 256, 256, 0, stream>>>(w, whi, wlo);
        elastic_mlp_wmma_kernel<true><<<grid, block, 0, stream>>>(x, w, whi, wlo, bb, out);
    } else {
        elastic_mlp_wmma_kernel<false><<<grid, block, 0, stream>>>(x, w, nullptr, nullptr, bb, out);
    }
    (void)n_in; (void)out_size;
}
